// DecoupledDynamicsModel_2688649527900
// MI455X (gfx1250) — compile-verified, matrix operand order checked by reading, not run on hardware
//
#include <hip/hip_runtime.h>
#include <hip/hip_bf16.h>
#include <stdint.h>
#include <math.h>

typedef unsigned short u16;
typedef __bf16 v16bf __attribute__((ext_vector_type(16)));
typedef float  v8f   __attribute__((ext_vector_type(8)));

#define NPOL   8      // P (outer and inner policy count)
#define ZDIM   64
#define DDIM   512
#define ADIM   8
#define HDIM   512
#define BATCH  8192
#define XK     96     // layer-1 K padded 72 -> 96 (3 steps of 32)
#define HPAD   520    // LDS row stride (u16) for hidden activations (bank spread)
#define NWAVES 8

union Frag {
  v16bf    bv;
  uint4    q[2];
  uint32_t w[8];
};

__device__ __forceinline__ u16 f2bf(float f) {
  union { float f; uint32_t u; } c; c.f = f;
  uint32_t x = c.u;
  return (u16)((x + 0x7fffu + ((x >> 16) & 1u)) >> 16);  // RNE
}

__device__ __forceinline__ v8f v8f_zero() {
  v8f z;
  #pragma unroll
  for (int i = 0; i < 8; ++i) z[i] = 0.0f;
  return z;
}

// ---------------- prep kernels ----------------

__global__ void dd_zero_meta(int* counts, int* offsets, int* cursors, int* tileStart) {
  int t = threadIdx.x;
  if (t < NPOL)     counts[t]   = 0;
  if (t < NPOL)     cursors[t]  = 0;
  if (t < NPOL + 1) offsets[t]  = 0;
  if (t < NPOL + 1) tileStart[t] = 0;
}

__global__ void dd_count(const int* __restrict__ pidx, int* __restrict__ counts) {
  int b = blockIdx.x * blockDim.x + threadIdx.x;
  if (b < BATCH) atomicAdd(&counts[pidx[b] & (NPOL - 1)], 1);
}

__global__ void dd_scan(const int* __restrict__ counts, int* __restrict__ offsets,
                        int* __restrict__ tileStart) {
  offsets[0] = 0; tileStart[0] = 0;
  for (int i = 0; i < NPOL; ++i) {
    offsets[i + 1]   = offsets[i] + counts[i];
    tileStart[i + 1] = tileStart[i] + (counts[i] + 15) / 16;
  }
}

__global__ void dd_scatter(const int* __restrict__ pidx, const int* __restrict__ offsets,
                           int* __restrict__ cursors, int* __restrict__ rowIdx) {
  int b = blockIdx.x * blockDim.x + threadIdx.x;
  if (b < BATCH) {
    int i = pidx[b] & (NPOL - 1);
    int pos = atomicAdd(&cursors[i], 1);
    rowIdx[offsets[i] + pos] = b;
  }
}

// Pack X = [latent_chunk(64) | action(8) | zeros(24)] as bf16, [B][NPOL][96]
__global__ void dd_pack_x(const float* __restrict__ lat, const float* __restrict__ act,
                          u16* __restrict__ Xbf) {
  int t = blockIdx.x * blockDim.x + threadIdx.x;
  if (t >= BATCH * NPOL * XK) return;
  int b = t / (NPOL * XK);
  int r = t - b * (NPOL * XK);
  int o = r / XK;
  int k = r - o * XK;
  float v = 0.0f;
  if (k < ZDIM)             v = lat[(size_t)b * DDIM + o * ZDIM + k];
  else if (k < ZDIM + ADIM) v = act[(size_t)b * ADIM + (k - ZDIM)];
  Xbf[t] = f2bf(v);
}

// Pre-swizzle a weight tensor [64 pairs][Ksrc][N] (row-major f32) into
// WMMA B-fragment order (bf16): per (pair, kstep, nchunk) fragment of 512
// elems = 32 lanes x 16 bf16, lane-contiguous (32B per lane).
// 16-bit B 32x16 layout assumption (mirrors the ISA 16-bit A layout):
// lane -> N column (l&15), half-wave g=l>>4 selects K+8 group; VGPR pairs hold
// consecutive K: K = kstep*32 + (e>=8?16:0) + g*8 + ((e>>1)&3)*2 + (e&1).
__global__ void dd_pack_w(const float* __restrict__ src, u16* __restrict__ dst,
                          int Ksrc, int N, int NC, int total) {
  int t = blockIdx.x * blockDim.x + threadIdx.x;
  if (t >= total) return;
  int f    = t >> 9;           // fragment id, flat over (pair, kstep, nchunk)
  int q    = t & 511;
  int lane = q >> 4;
  int e    = q & 15;
  int KS = (total / (64 * 512)) / NC;
  int pairFrags = KS * NC;
  int pair  = f / pairFrags;
  int fr    = f - pair * pairFrags;
  int kstep = fr / NC;
  int n     = fr - kstep * NC;
  int g     = lane >> 4;
  int ncol  = n * 16 + (lane & 15);
  int K = kstep * 32 + ((e >> 3) << 4) + (g << 3) + (((e >> 1) & 3) << 1) + (e & 1);
  float v = (K < Ksrc) ? src[((size_t)pair * Ksrc + K) * N + ncol] : 0.0f;
  dst[t] = f2bf(v);
}

// ---------------- fused 3-layer MLP ----------------
// grid.x = tile slots (<= 520). Block = 8 waves; wave w = inner policy w.
// Per wave: 16 rows x (96->512 relu ->512 relu ->64), staging h1/h2 in LDS.
__global__ __launch_bounds__(256, 1)
void dd_mlp(const u16* __restrict__ Xbf,
            const u16* __restrict__ W1p, const u16* __restrict__ W2p,
            const u16* __restrict__ W3p,
            const float* __restrict__ b1, const float* __restrict__ b2,
            const float* __restrict__ b3,
            const int* __restrict__ rowIdx, const int* __restrict__ counts,
            const int* __restrict__ offsets, const int* __restrict__ tileStart,
            float* __restrict__ out) {
  __shared__ int rowsS[16];
  __shared__ int polS, tS, cntS;
  extern __shared__ u16 hbuf[];

  int bx = blockIdx.x;
  if (threadIdx.x == 0) {
    int total = tileStart[NPOL];
    if (bx >= total) {
      polS = -1;
    } else {
      int i = 0;
      while (i < NPOL - 1 && bx >= tileStart[i + 1]) ++i;
      polS = i;
      tS   = bx - tileStart[i];
      cntS = counts[i];
    }
  }
  __syncthreads();
  int pol = polS;
  if (pol < 0) return;
  if (threadIdx.x < 16) {
    int k = tS * 16 + threadIdx.x;
    int idx = offsets[pol] + (k < cntS ? k : 0);   // clamp tail to a valid row
    rowsS[threadIdx.x] = rowIdx[idx];
  }
  __syncthreads();

  int lane = threadIdx.x & 31;
  int p    = threadIdx.x >> 5;      // inner policy = wave id
  int m    = lane & 15;             // matrix row / column within tile
  int g    = lane >> 4;             // half-wave K-group select
  int pair = pol * NPOL + p;

  u16* h1 = hbuf + (size_t)p * (2 * 16 * HPAD);
  u16* h2 = h1 + 16 * HPAD;

  // ---- Layer 1: X[16,96] @ W1[96,512] + b1, ReLU -> h1 (bf16 in LDS) ----
  {
    const u16* xrow = Xbf + ((size_t)rowsS[m] * NPOL + p) * XK;
    Frag a1[3];                                    // hoist all A-fragments once
    #pragma unroll
    for (int k = 0; k < 3; ++k) {
      #pragma unroll
      for (int j = 0; j < 8; ++j) {
        int kp = (k << 5) + ((j >> 2) << 4) + (g << 3) + ((j & 3) << 1);
        a1[k].w[j] = *(const uint32_t*)(xrow + kp);
      }
    }
    const u16* w1   = W1p + (size_t)pair * (3 * 32 * 512);
    const float* bb = b1 + (size_t)pair * HDIM;
    for (int nb = 0; nb < 4; ++nb) {               // 8 N-chunks per block
      v8f acc[8];
      #pragma unroll
      for (int j = 0; j < 8; ++j) acc[j] = v8f_zero();
      for (int k = 0; k < 3; ++k) {
        #pragma unroll
        for (int j = 0; j < 8; ++j) {
          Frag bq;
          const uint4* bp = (const uint4*)(w1 + ((size_t)(k * 32 + nb * 8 + j) << 9) + (lane << 4));
          bq.q[0] = bp[0]; bq.q[1] = bp[1];
          acc[j] = __builtin_amdgcn_wmma_f32_16x16x32_bf16(
              false, a1[k].bv, false, bq.bv, (short)0, acc[j], false, false);
        }
      }
      #pragma unroll
      for (int j = 0; j < 8; ++j) {
        int n = nb * 8 + j;
        float bias = bb[n * 16 + m];               // C/D: column = lane&15
        #pragma unroll
        for (int r = 0; r < 8; ++r) {
          int M = r + (g << 3);                    // C/D: row = vgpr + 8*half
          float v = fmaxf(acc[j][r] + bias, 0.0f);
          h1[M * HPAD + n * 16 + m] = f2bf(v);
        }
      }
    }
  }

  // ---- Layer 2: h1[16,512] @ W2[512,512] + b2, ReLU -> h2 ----
  {
    const u16* w2   = W2p + (size_t)pair * (16 * 32 * 512);
    const float* bb = b2 + (size_t)pair * HDIM;
    for (int nb = 0; nb < 4; ++nb) {               // 8 N-chunks per block
      v8f acc[8];
      #pragma unroll
      for (int j = 0; j < 8; ++j) acc[j] = v8f_zero();
      for (int k = 0; k < 16; ++k) {
        if (k + 1 < 16)
          __builtin_prefetch(w2 + ((size_t)((k + 1) * 32 + nb * 8) << 9) + (lane << 4), 0, 0);
        Frag a;
        #pragma unroll
        for (int j = 0; j < 8; ++j) {
          int kp = (k << 5) + ((j >> 2) << 4) + (g << 3) + ((j & 3) << 1);
          a.w[j] = *(const uint32_t*)(h1 + m * HPAD + kp);
        }
        #pragma unroll
        for (int j = 0; j < 8; ++j) {
          Frag bq;
          const uint4* bp = (const uint4*)(w2 + ((size_t)(k * 32 + nb * 8 + j) << 9) + (lane << 4));
          bq.q[0] = bp[0]; bq.q[1] = bp[1];
          acc[j] = __builtin_amdgcn_wmma_f32_16x16x32_bf16(
              false, a.bv, false, bq.bv, (short)0, acc[j], false, false);
        }
      }
      #pragma unroll
      for (int j = 0; j < 8; ++j) {
        int n = nb * 8 + j;
        float bias = bb[n * 16 + m];
        #pragma unroll
        for (int r = 0; r < 8; ++r) {
          int M = r + (g << 3);
          float v = fmaxf(acc[j][r] + bias, 0.0f);
          h2[M * HPAD + n * 16 + m] = f2bf(v);
        }
      }
    }
  }

  // ---- Layer 3: h2[16,512] @ W3[512,64] + b3 -> out (f32, scattered) ----
  {
    const u16* w3   = W3p + (size_t)pair * (16 * 4 * 512);
    const float* bb = b3 + (size_t)pair * ZDIM;
    v8f acc[4];
    #pragma unroll
    for (int j = 0; j < 4; ++j) acc[j] = v8f_zero();
    for (int k = 0; k < 16; ++k) {
      Frag a;
      #pragma unroll
      for (int j = 0; j < 8; ++j) {
        int kp = (k << 5) + ((j >> 2) << 4) + (g << 3) + ((j & 3) << 1);
        a.w[j] = *(const uint32_t*)(h2 + m * HPAD + kp);
      }
      #pragma unroll
      for (int j = 0; j < 4; ++j) {
        Frag bq;
        const uint4* bp = (const uint4*)(w3 + ((size_t)(k * 4 + j) << 9) + (lane << 4));
        bq.q[0] = bp[0]; bq.q[1] = bp[1];
        acc[j] = __builtin_amdgcn_wmma_f32_16x16x32_bf16(
            false, a.bv, false, bq.bv, (short)0, acc[j], false, false);
      }
    }
    int cnt = cntS, t0 = tS * 16;
    #pragma unroll
    for (int j = 0; j < 4; ++j) {
      float bias = bb[j * 16 + m];
      #pragma unroll
      for (int r = 0; r < 8; ++r) {
        int M = r + (g << 3);
        if (t0 + M < cnt)
          out[(size_t)rowsS[M] * DDIM + p * ZDIM + j * 16 + m] = acc[j][r] + bias;
      }
    }
  }
}

// ---------------- host launcher ----------------

extern "C" void kernel_launch(void* const* d_in, const int* in_sizes, int n_in,
                              void* d_out, int out_size, void* d_ws, size_t ws_size,
                              hipStream_t stream) {
  const float* latents = (const float*)d_in[0];
  const float* actions = (const float*)d_in[1];
  const int*   pidx    = (const int*)d_in[2];
  const float* W1 = (const float*)d_in[3];
  const float* b1 = (const float*)d_in[4];
  const float* W2 = (const float*)d_in[5];
  const float* b2 = (const float*)d_in[6];
  const float* W3 = (const float*)d_in[7];
  const float* b3 = (const float*)d_in[8];
  float* out = (float*)d_out;

  char* ws = (char*)d_ws;
  int* counts    = (int*)(ws + 0);
  int* offsets   = (int*)(ws + 64);
  int* cursors   = (int*)(ws + 128);
  int* tileStart = (int*)(ws + 192);
  int* rowIdx    = (int*)(ws + 256);

  constexpr size_t O_XBF  = 256 + (size_t)BATCH * 4;                   // 33024
  constexpr size_t SZ_XBF = (size_t)BATCH * NPOL * XK * 2;             // 12.6 MB
  constexpr size_t SZ_W1P = (size_t)64 * 3 * 32 * 512 * 2;             // 6.3 MB
  constexpr size_t SZ_W2P = (size_t)64 * 16 * 32 * 512 * 2;            // 33.5 MB
  constexpr size_t O_W1P = O_XBF + SZ_XBF;
  constexpr size_t O_W2P = O_W1P + SZ_W1P;
  constexpr size_t O_W3P = O_W2P + SZ_W2P;
  u16* Xbf = (u16*)(ws + O_XBF);
  u16* W1p = (u16*)(ws + O_W1P);
  u16* W2p = (u16*)(ws + O_W2P);
  u16* W3p = (u16*)(ws + O_W3P);

  // bucket rows by outer policy
  hipLaunchKernelGGL(dd_zero_meta, dim3(1), dim3(64), 0, stream,
                     counts, offsets, cursors, tileStart);
  hipLaunchKernelGGL(dd_count, dim3(BATCH / 256), dim3(256), 0, stream, pidx, counts);
  hipLaunchKernelGGL(dd_scan, dim3(1), dim3(1), 0, stream, counts, offsets, tileStart);
  hipLaunchKernelGGL(dd_scatter, dim3(BATCH / 256), dim3(256), 0, stream,
                     pidx, offsets, cursors, rowIdx);

  // bf16 packing / weight swizzle
  hipLaunchKernelGGL(dd_pack_x, dim3((BATCH * NPOL * XK) / 256), dim3(256), 0, stream,
                     latents, actions, Xbf);
  hipLaunchKernelGGL(dd_pack_w, dim3((64 * 3 * 32 * 512) / 256), dim3(256), 0, stream,
                     W1, W1p, 72, 512, 32, 64 * 3 * 32 * 512);
  hipLaunchKernelGGL(dd_pack_w, dim3((64 * 16 * 32 * 512) / 256), dim3(256), 0, stream,
                     W2, W2p, 512, 512, 32, 64 * 16 * 32 * 512);
  hipLaunchKernelGGL(dd_pack_w, dim3((64 * 16 * 4 * 512) / 256), dim3(256), 0, stream,
                     W3, W3p, 512, 64, 4, 64 * 16 * 4 * 512);

  // fused MLP: <=520 row tiles, 8 waves (inner policies) per block,
  // 266 KB dynamic LDS (h1+h2 per wave) on the 320 KB WGP.
  constexpr int MAX_TILES = BATCH / 16 + NPOL;                          // 520
  constexpr size_t LDS_BYTES = (size_t)NWAVES * 2 * 16 * HPAD * sizeof(u16);
  hipLaunchKernelGGL(dd_mlp, dim3(MAX_TILES), dim3(256), LDS_BYTES, stream,
                     Xbf, W1p, W2p, W3p, b1, b2, b3,
                     rowIdx, counts, offsets, tileStart, out);
}
